// BSGRUv2_5437428596859
// MI455X (gfx1250) — matrix operand order repaired and checked by reading, not compiled, not run on hardware
//
#include <hip/hip_runtime.h>
#include <hip/hip_bf16.h>

// Problem constants (from reference)
#define T_   512
#define B_   64
#define I_   256
#define H_   512
#define H3_  1536
#define NB_  8
#define NWG_ 16      // workgroups in the persistent scan (one per 32-col H slice)

// Scan-kernel LDS layout (bytes)
#define LDS_A_OFF    0         // 64x512 bf16          = 65536
#define LDS_U_OFF    65536     // 96x512 bf16          = 98304
#define LDS_HALL_OFF 163840    // 64x96  f32           = 24576
#define LDS_WX_OFF   188416    // 2 x (64x3x32) f32    = 49152
#define LDS_K_OFF    237568    // ky + kx (64x8 f32)x2 =  4096
#define LDS_TOTAL    241664

typedef __bf16 bf16_t;
typedef __attribute__((ext_vector_type(8)))  bf16_t bf16x8;
typedef __attribute__((ext_vector_type(16))) bf16_t bf16x16;
typedef __attribute__((ext_vector_type(8)))  float  f32x8;
typedef __attribute__((ext_vector_type(4)))  unsigned u32x4;
typedef __attribute__((ext_vector_type(8)))  unsigned u32x8;

__device__ __forceinline__ bf16_t f2bf(float f) {
  unsigned u = __builtin_bit_cast(unsigned, f);
  unsigned r = u + 0x7FFFu + ((u >> 16) & 1u);   // round-to-nearest-even
  return __builtin_bit_cast(bf16_t, (unsigned short)(r >> 16));
}

__device__ __forceinline__ f32x8 wmma_bf16(bf16x16 a, bf16x16 b, f32x8 c) {
  return __builtin_amdgcn_wmma_f32_16x16x32_bf16(false, a, false, b, (short)0, c,
                                                 false, false);
}

// A-fragment (16x32 bf16, row-major source, row = lane&15):
// lane group g=(lane>>4) holds K = {16g..16g+15} (contiguous, matches B)
__device__ __forceinline__ bf16x16 make_a_frag(const bf16_t* row, int g) {
  bf16x8 lo = *(const bf16x8*)(row + 16 * g);
  bf16x8 hi = *(const bf16x8*)(row + 16 * g + 8);
  bf16x16 a;
#pragma unroll
  for (int i = 0; i < 8; ++i) { a[i] = lo[i]; a[8 + i] = hi[i]; }
  return a;
}

// Low 32 bits of a generic pointer to __shared__ == LDS byte offset
__device__ __forceinline__ unsigned lds_off(const void* p) {
  return (unsigned)(unsigned long long)p;
}

// Tensor Data Mover: 3D tile load Global->LDS.  Descriptor per ISA ch.8:
//  group0: [1:0]=count(1), [63:32]=lds_addr, [120:64]=global_addr, [127:126]=2
//  group1: [17:16]=data_size, [79:48]=tensor_dim0, [111:80]=tensor_dim1,
//          [127:112]=tile0, [143:128]=tile1, [159:144]=tile2,
//          [207:160]=dim0_stride, [255:208]=dim1_stride   (wg_mask=0: no cluster)
//  group2: [31:0]=tensor_dim2 ; group3: zero (no dim3/dim4)
__device__ __forceinline__ void tdm_load_3d(
    unsigned lds_addr, const void* gptr, unsigned ds_code,
    unsigned td0, unsigned td1, unsigned td2,
    unsigned long long st0, unsigned long long st1,
    unsigned tl0, unsigned tl1, unsigned tl2) {
  unsigned long long ga = (unsigned long long)gptr;
  u32x4 g0;
  g0[0] = 1u;                                            // count=1, user mode
  g0[1] = lds_addr;
  g0[2] = (unsigned)(ga & 0xffffffffu);
  g0[3] = (unsigned)((ga >> 32) & 0x01ffffffu) | (2u << 30);   // type=2 (image)
  u32x8 g1;
  g1[0] = (ds_code & 3u) << 16;
  g1[1] = (td0 & 0xffffu) << 16;
  g1[2] = (td0 >> 16) | ((td1 & 0xffffu) << 16);
  g1[3] = (td1 >> 16) | ((tl0 & 0xffffu) << 16);
  g1[4] = (tl1 & 0xffffu) | ((tl2 & 0xffffu) << 16);
  g1[5] = (unsigned)(st0 & 0xffffffffu);
  g1[6] = (unsigned)((st0 >> 32) & 0xffffu) | ((unsigned)(st1 & 0xffffu) << 16);
  g1[7] = (unsigned)((st1 >> 16) & 0xffffffffu);
  u32x4 g2; g2[0] = td2; g2[1] = 0u; g2[2] = 0u; g2[3] = 0u;
  u32x4 g3; g3[0] = 0u;  g3[1] = 0u; g3[2] = 0u; g3[3] = 0u;
  asm volatile("tensor_load_to_lds %0, %1, %2, %3"
               :: "s"(g0), "s"(g1), "s"(g2), "s"(g3) : "memory");
}

// ---------------------------------------------------------------------------
__global__ void cvt_bf16_kernel(const float* __restrict__ src,
                                bf16_t* __restrict__ dst, int n) {
  int i = blockIdx.x * blockDim.x + threadIdx.x;
  if (i < n) dst[i] = f2bf(src[i]);
}

// ---------------------------------------------------------------------------
// Kx[t,b,n] = in[t,b,:] . K_w[n,:I] + K_b[n]
__global__ void kx_kernel(const float* __restrict__ x,
                          const float* __restrict__ Kw,
                          const float* __restrict__ Kb,
                          float* __restrict__ Kx) {
  int idx = blockIdx.x * blockDim.x + threadIdx.x;
  if (idx >= T_ * B_ * NB_) return;
  int n = idx & (NB_ - 1);
  int tb = idx >> 3;
  const float4* x4 = (const float4*)(x + (size_t)tb * I_);
  const float4* k4 = (const float4*)(Kw + (size_t)n * (I_ + H_));
  float acc = 0.f;
#pragma unroll 8
  for (int i = 0; i < I_ / 4; ++i) {
    float4 a = x4[i], c = k4[i];
    acc += a.x * c.x + a.y * c.y + a.z * c.z + a.w * c.w;
  }
  Kx[idx] = acc + Kb[n];
}

// ---------------------------------------------------------------------------
// Wx = X(32768x256) * W_w^T(256x1536) + W_b, bf16 WMMA, one 16x16 tile / wave
__global__ __launch_bounds__(256) void wx_gemm_kernel(
    const bf16_t* __restrict__ X16, const bf16_t* __restrict__ Ww16,
    const float* __restrict__ Wbias, float* __restrict__ Wx) {
  const int lane = threadIdx.x & 31;
  const int wave = threadIdx.x >> 5;
  const int tile = blockIdx.x * 8 + wave;
  const int nt = tile % (H3_ / 16);
  const int mt = tile / (H3_ / 16);
  const int m0 = mt * 16, n0 = nt * 16;

  const int ag = (lane >> 4) & 1;
  const bf16_t* arow = X16 + (size_t)(m0 + (lane & 15)) * I_;
  const bf16_t* brow = Ww16 + (size_t)(n0 + (lane & 15)) * I_;

  f32x8 acc = {};
#pragma unroll
  for (int k0 = 0; k0 < I_; k0 += 32) {
    bf16x16 af = make_a_frag(arow + k0, ag);
    bf16x16 bfv = *(const bf16x16*)(brow + k0 + 16 * ag);
    acc = wmma_bf16(af, bfv, acc);
  }
  const int n = lane & 15, g = lane >> 4;
  const float bias = Wbias[n0 + n];
#pragma unroll
  for (int v = 0; v < 8; ++v) {
    int m = m0 + g * 8 + v;
    Wx[(size_t)m * H3_ + n0 + n] = acc[v] + bias;
  }
}

// ---------------------------------------------------------------------------
__global__ void scan_init_kernel(float* __restrict__ hbuf0,
                                 unsigned* __restrict__ bar) {
  int i = blockIdx.x * blockDim.x + threadIdx.x;
  if (i < B_ * H_) hbuf0[i] = 0.f;
  if (i == 0) *bar = 0u;
}

// ---------------------------------------------------------------------------
// Persistent scan: NWG_ workgroups, global spin-barrier per timestep.
// WG wg owns H columns [wg*32, wg*32+32) of all three gates.
// U_w slice: TDM-staged to LDS once.  Wx[t] slice: TDM double-buffered, one
// step ahead, synchronized via s_wait_tensorcnt.
__global__ __launch_bounds__(256) void scan_kernel(
    const float* __restrict__ Wx,    // (T,B,1536)
    const float* __restrict__ Kx,    // (T,B,8)
    const float* __restrict__ Kw,    // (8,768): cols [256,768) are K_h
    const bf16_t* __restrict__ U16,  // (1536,512) bf16 row-major
    const float* __restrict__ Ub,    // (1536)
    float* __restrict__ hbuf,        // (2,64,512) ping-pong hidden state
    unsigned* __restrict__ bar,
    float* __restrict__ out_seq,     // (T,B,512)
    float* __restrict__ out_blk) {   // (T,B,8)
  extern __shared__ char smem[];
  bf16_t* sh_A    = (bf16_t*)(smem + LDS_A_OFF);     // [64][512]
  bf16_t* sh_U    = (bf16_t*)(smem + LDS_U_OFF);     // [96][512] (gate*32+r)
  float*  sh_hall = (float*)(smem + LDS_HALL_OFF);   // [64][96]
  float*  sh_wx   = (float*)(smem + LDS_WX_OFF);     // [2][64][3][32]
  float*  sh_ky   = (float*)(smem + LDS_K_OFF);      // [64][8]
  float*  sh_kx   = sh_ky + B_ * NB_;

  const int tid  = threadIdx.x;
  const int lane = tid & 31;
  const int wave = tid >> 5;
  const int wg   = blockIdx.x;
  const int c0   = wg * 32;

  if (tid < B_ * NB_) sh_kx[tid] = 0.f;   // k0 = zeros

  // --- one-time TDM stage of this WG's U_w slice: tile x=512,y=32 rows,z=3 gates
  if (wave == 0) {
    tdm_load_3d(lds_off(sh_U), U16 + (size_t)c0 * H_, /*bf16*/1u,
                512u, 32u, 3u,
                512ull /*row*/, (unsigned long long)H_ * 512ull /*gate*/,
                512u, 32u, 3u);
    // prefetch Wx tile for t=0: x=32 cols, y=3 gates (stride 512), z=64 b (stride 1536)
    tdm_load_3d(lds_off(sh_wx), Wx + c0, /*f32*/2u,
                512u, 3u, 64u, 512ull, 1536ull, 32u, 3u, 64u);
  }
  __syncthreads();

  for (int t = 0; t < T_; ++t) {
    const float* hx  = hbuf + (size_t)(t & 1) * (B_ * H_);
    float*       hy  = hbuf + (size_t)((t + 1) & 1) * (B_ * H_);
    const float* kxt = Kx + (size_t)t * B_ * NB_;
    const float* swx = sh_wx + (size_t)(t & 1) * (B_ * 96);

    // --- 0) prefetch next step's Wx tile; wait for current tile (in-order TDM)
    if (wave == 0) {
      if (t + 1 < T_) {
        tdm_load_3d(lds_off(sh_wx + (size_t)((t + 1) & 1) * (B_ * 96)),
                    Wx + (size_t)(t + 1) * B_ * H3_ + c0, 2u,
                    512u, 3u, 64u, 512ull, 1536ull, 32u, 3u, 64u);
        __builtin_amdgcn_s_wait_tensorcnt(1);   // t-tile (and U) complete
      } else {
        __builtin_amdgcn_s_wait_tensorcnt(0);
      }
    }
    __syncthreads();

    // --- 1) logits[b,n] = tau*(Kx + hx[b,:].K_h[n,:])
    for (int p = tid; p < B_ * NB_; p += 256) {
      int b = p >> 3, n = p & 7;
      const float4* h4 = (const float4*)(hx + (size_t)b * H_);
      const float4* k4 = (const float4*)(Kw + (size_t)n * (I_ + H_) + I_);
      float acc = 0.f;
#pragma unroll 8
      for (int i = 0; i < H_ / 4; ++i) {
        float4 a = h4[i], c = k4[i];
        acc += a.x * c.x + a.y * c.y + a.z * c.z + a.w * c.w;
      }
      sh_ky[p] = 5.0f * (kxt[p] + acc);
    }
    __syncthreads();

    // --- 2) softmax over NB per batch row, in place
    if (tid < B_) {
      float l[NB_], m = -1e30f;
#pragma unroll
      for (int n = 0; n < NB_; ++n) { l[n] = sh_ky[tid * NB_ + n]; m = fmaxf(m, l[n]); }
      float s = 0.f;
#pragma unroll
      for (int n = 0; n < NB_; ++n) { l[n] = __expf(l[n] - m); s += l[n]; }
      float inv = 1.f / s;
#pragma unroll
      for (int n = 0; n < NB_; ++n) sh_ky[tid * NB_ + n] = l[n] * inv;
    }

    // --- 3) stage A = hx * repeat(kx_prev, 64) as bf16 into LDS (64x512)
    for (int e = tid; e < B_ * H_; e += 256) {
      int b = e >> 9, k = e & (H_ - 1);
      sh_A[e] = f2bf(hx[e] * sh_kx[b * NB_ + (k >> 6)]);
    }
    __syncthreads();

    // --- 4) h_all_raw = A(64x512) @ U_slice^T(512x96) via bf16 WMMA, all LDS
    for (int tt = wave; tt < 24; tt += 8) {
      int mt = tt & 3, nt = tt >> 2;
      int m0 = mt * 16;
      int ru = (nt >> 1) * 32 + (nt & 1) * 16 + (lane & 15);   // sh_U row
      const bf16_t* brow = sh_U + (size_t)ru * H_;
      const bf16_t* arow = sh_A + (size_t)(m0 + (lane & 15)) * H_;
      const int ag = (lane >> 4) & 1;
      f32x8 acc = {};
#pragma unroll
      for (int k0 = 0; k0 < H_; k0 += 32) {
        bf16x16 af  = make_a_frag(arow + k0, ag);
        bf16x16 bfv = *(const bf16x16*)(brow + k0 + 16 * ag);
        acc = wmma_bf16(af, bfv, acc);
      }
      const int n = lane & 15, g = lane >> 4;
#pragma unroll
      for (int v = 0; v < 8; ++v)
        sh_hall[(m0 + g * 8 + v) * 96 + nt * 16 + n] = acc[v];
    }
    __syncthreads();

    // --- 5) gates + state update; Wx read from TDM-staged LDS [b][gate][32]
    for (int e = tid; e < B_ * 32; e += 256) {
      int b = e >> 5, hh = e & 31;
      int h = c0 + hh;
      float ky = sh_ky[b * NB_ + (h >> 6)];
      const float* wr = swx + b * 96;
      float iz  = wr[hh]      * ky;
      float ir  = wr[32 + hh] * ky;
      float in_ = wr[64 + hh] * ky;
      const float* hl = sh_hall + b * 96;
      float hz = (hl[hh]      + Ub[h])          * ky;
      float hr = (hl[32 + hh] + Ub[H_ + h])     * ky;
      float hn = (hl[64 + hh] + Ub[2 * H_ + h]) * ky;
      float z  = 1.f / (1.f + __expf(-(iz + hz)));
      float r  = 1.f / (1.f + __expf(-(ir + hr)));
      float nn = tanhf(in_ + hn * r);
      float hprev = hx[b * H_ + h];
      float hv = nn + z * (hprev - nn);
      hy[b * H_ + h] = hv;
      out_seq[((size_t)t * B_ + b) * H_ + h] = hv;
    }
    if (wg == 0 && tid < B_ * NB_)
      out_blk[(size_t)t * B_ * NB_ + tid] = sh_ky[tid];
    __syncthreads();

    // --- 6) carry kx <- ky
    if (tid < B_ * NB_) sh_kx[tid] = sh_ky[tid];

    // --- 7) grid-wide barrier: all hy slices visible before next step
    __threadfence();
    __syncthreads();
    if (tid == 0) {
      __hip_atomic_fetch_add(bar, 1u, __ATOMIC_RELEASE, __HIP_MEMORY_SCOPE_AGENT);
      const unsigned target = (unsigned)NWG_ * (unsigned)(t + 1);
      while (__hip_atomic_load(bar, __ATOMIC_ACQUIRE, __HIP_MEMORY_SCOPE_AGENT) < target)
        __builtin_amdgcn_s_sleep(2);
    }
    __syncthreads();
  }
}

// ---------------------------------------------------------------------------
extern "C" void kernel_launch(void* const* d_in, const int* in_sizes, int n_in,
                              void* d_out, int out_size, void* d_ws, size_t ws_size,
                              hipStream_t stream) {
  const float* in_tensor = (const float*)d_in[0];
  const float* K_w = (const float*)d_in[1];
  const float* K_b = (const float*)d_in[2];
  const float* W_w = (const float*)d_in[3];
  const float* W_b = (const float*)d_in[4];
  const float* U_w = (const float*)d_in[5];
  const float* U_b = (const float*)d_in[6];

  float* out_seq = (float*)d_out;
  float* out_blk = out_seq + (size_t)T_ * B_ * H_;

  char* ws = (char*)d_ws;
  float*  Wx   = (float*)(ws);                          // 201,326,592
  bf16_t* X16  = (bf16_t*)(ws + 201326592);             //  16,777,216
  bf16_t* Ww16 = (bf16_t*)(ws + 218103808);             //     786,432
  bf16_t* Uw16 = (bf16_t*)(ws + 218890240);             //   1,572,864
  float*  Kx   = (float*)(ws + 220463104);              //   1,048,576
  float*  hbuf = (float*)(ws + 221511680);              //     262,144
  unsigned* bar = (unsigned*)(ws + 221773824);

  cvt_bf16_kernel<<<(T_ * B_ * I_ + 255) / 256, 256, 0, stream>>>(in_tensor, X16, T_ * B_ * I_);
  cvt_bf16_kernel<<<(H3_ * I_ + 255) / 256, 256, 0, stream>>>(W_w, Ww16, H3_ * I_);
  cvt_bf16_kernel<<<(H3_ * H_ + 255) / 256, 256, 0, stream>>>(U_w, Uw16, H3_ * H_);

  kx_kernel<<<(T_ * B_ * NB_ + 255) / 256, 256, 0, stream>>>(in_tensor, K_w, K_b, Kx);

  wx_gemm_kernel<<<(2048 * 96) / 8, 256, 0, stream>>>(X16, Ww16, W_b, Wx);

  scan_init_kernel<<<(B_ * H_ + 255) / 256, 256, 0, stream>>>(hbuf, bar);

  scan_kernel<<<NWG_, 256, LDS_TOTAL, stream>>>(Wx, Kx, K_w, Uw16, U_b,
                                                hbuf, bar, out_seq, out_blk);
}
